// Net_3582002725505
// MI455X (gfx1250) — compile-verified, matrix operand order checked by reading, not run on hardware
//
#include <hip/hip_runtime.h>
#include <hip/hip_bf16.h>

typedef __attribute__((ext_vector_type(16))) _Float16 v16h;
typedef __attribute__((ext_vector_type(8)))  float    v8f;
typedef __attribute__((ext_vector_type(8)))  int      v8i;
typedef __attribute__((ext_vector_type(4)))  int      i32x4;
typedef __attribute__((ext_vector_type(2)))  int      i32x2;

union Frag16 { i32x4 q[2]; v16h h; };
union Frag8  { i32x2 d[4]; v8i  v; };

#if defined(__has_builtin)
#if __has_builtin(__builtin_amdgcn_global_load_async_to_lds_b128) && \
    __has_builtin(__builtin_amdgcn_s_wait_asynccnt)
#define USE_ASYNC_LDS 1
#endif
#endif
#ifndef USE_ASYNC_LDS
#define USE_ASYNC_LDS 0
#endif

typedef __attribute__((address_space(1))) i32x4 as1_i32x4;
typedef __attribute__((address_space(3))) i32x4 as3_i32x4;

#if USE_ASYNC_LDS
#define ASYNC_COPY16(ldsp, gp)                                              \
    __builtin_amdgcn_global_load_async_to_lds_b128(                         \
        (as1_i32x4*)(void*)(gp), (as3_i32x4*)(void*)(ldsp), 0, 0)
#define ASYNC_WAIT() __builtin_amdgcn_s_wait_asynccnt(0)
#else
#define ASYNC_COPY16(ldsp, gp) (*(i32x4*)(ldsp) = *(const i32x4*)(gp))
#define ASYNC_WAIT() ((void)0)
#endif

__global__ void cvt_pad_f16_kernel(const float* __restrict__ in,
                                   _Float16* __restrict__ out,
                                   int rows, int inc, int outc, int binarize)
{
    int idx = blockIdx.x * blockDim.x + threadIdx.x;
    int total = rows * outc;
    if (idx >= total) return;
    int r = idx / outc;
    int c = idx - r * outc;
    float v = 0.0f;
    if (c < inc) {
        v = in[(size_t)r * inc + c];
        if (binarize) v = (v >= 0.0f) ? 1.0f : -1.0f;
    }
    out[idx] = (_Float16)v;
}

__global__ void sign_i8_kernel(const float* __restrict__ in,
                               signed char* __restrict__ out, int n)
{
    int idx = blockIdx.x * blockDim.x + threadIdx.x;
    if (idx >= n) return;
    out[idx] = (in[idx] >= 0.0f) ? (signed char)1 : (signed char)-1;
}

__global__ __launch_bounds__(256) void gemm_f16_kernel(
    const _Float16* __restrict__ A,
    const _Float16* __restrict__ Bm,
    const float* __restrict__ bias, const float* __restrict__ gam,
    const float* __restrict__ bet,  const float* __restrict__ mean,
    const float* __restrict__ var,
    signed char* __restrict__ out,
    int N, int K)
{
    alignas(16) __shared__ char AsB[2][128 * 64];
    alignas(16) __shared__ char BsB[2][128 * 64];

    const int tid  = threadIdx.x;
    const int wave = tid >> 5;
    const int lane = tid & 31;
    const int r16  = lane & 15;
    const int hi   = lane >> 4;
    const int wm   = wave >> 2;
    const int wn   = wave & 3;
    const int blockN = blockIdx.x * 128;
    const int blockM = blockIdx.y * 128;

    const char* Ag = (const char*)(A  + (size_t)blockM * K);
    const char* Bg = (const char*)(Bm + (size_t)blockN * K);
    const size_t rowB = (size_t)K * 2;

    const int c0   = tid;
    const int row0 = c0 >> 2, col0 = (c0 & 3) << 4;
    const int row1 = (c0 + 256) >> 2, col1 = ((c0 + 256) & 3) << 4;

    auto issue_tile = [&](int kt, int buf) {
        const size_t kb = (size_t)kt * 2;
        ASYNC_COPY16(&AsB[buf][row0 * 64 + col0], Ag + (size_t)row0 * rowB + kb + col0);
        ASYNC_COPY16(&BsB[buf][row0 * 64 + col0], Bg + (size_t)row0 * rowB + kb + col0);
        ASYNC_COPY16(&AsB[buf][row1 * 64 + col1], Ag + (size_t)row1 * rowB + kb + col1);
        ASYNC_COPY16(&BsB[buf][row1 * 64 + col1], Bg + (size_t)row1 * rowB + kb + col1);
    };

    v8f acc[4][2] = {};
    const int nt = K / 32;

    issue_tile(0, 0);
    for (int it = 0; it < nt; ++it) {
        const int cur = it & 1;
        ASYNC_WAIT();
        __syncthreads();
        if (it + 1 < nt) issue_tile((it + 1) * 32, cur ^ 1);

        const char* As = AsB[cur];
        const char* Bs = BsB[cur];
        Frag16 bfr[2];
#pragma unroll
        for (int nf = 0; nf < 2; ++nf) {
            int br = wn * 32 + nf * 16 + r16;
            bfr[nf].q[0] = *(const i32x4*)(Bs + br * 64 + hi * 16);
            bfr[nf].q[1] = *(const i32x4*)(Bs + br * 64 + 32 + hi * 16);
        }
#pragma unroll
        for (int mf = 0; mf < 4; ++mf) {
            Frag16 a;
            int ar = wm * 64 + mf * 16 + r16;
            a.q[0] = *(const i32x4*)(As + ar * 64 + hi * 16);
            a.q[1] = *(const i32x4*)(As + ar * 64 + 32 + hi * 16);
#pragma unroll
            for (int nf = 0; nf < 2; ++nf) {
                acc[mf][nf] = __builtin_amdgcn_wmma_f32_16x16x32_f16(
                    false, a.h, false, bfr[nf].h, (short)0, acc[mf][nf],
                    false, false);
            }
        }
    }

#pragma unroll
    for (int nf = 0; nf < 2; ++nf) {
        const int n = blockN + wn * 32 + nf * 16 + r16;
        const float rs = rsqrtf(var[n] + 1e-5f);
        const float s  = gam[n] * rs;
        const float t  = s * (bias[n] - mean[n]) + bet[n];
#pragma unroll
        for (int mf = 0; mf < 4; ++mf) {
            const int mbase = blockM + wm * 64 + mf * 16 + hi * 8;
#pragma unroll
            for (int r = 0; r < 8; ++r) {
                float v = s * acc[mf][nf][r] + t;
                v = fminf(1.0f, fmaxf(-1.0f, v));
                out[(size_t)(mbase + r) * N + n] =
                    (v >= 0.0f) ? (signed char)1 : (signed char)-1;
            }
        }
    }
}

template <int BINOUT>
__global__ __launch_bounds__(256) void gemm_i8_kernel(
    const signed char* __restrict__ A,
    const signed char* __restrict__ Bm,
    const float* __restrict__ bias, const float* __restrict__ gam,
    const float* __restrict__ bet,  const float* __restrict__ mean,
    const float* __restrict__ var,
    signed char* __restrict__ out8,
    float* __restrict__ outf,
    int N, int K)
{
    alignas(16) __shared__ char AsB[2][128 * 64];
    alignas(16) __shared__ char BsB[2][128 * 64];

    const int tid  = threadIdx.x;
    const int wave = tid >> 5;
    const int lane = tid & 31;
    const int r16  = lane & 15;
    const int hi   = lane >> 4;
    const int wm   = wave >> 2;
    const int wn   = wave & 3;
    const int blockN = blockIdx.x * 128;
    const int blockM = blockIdx.y * 128;

    const char* Ag = (const char*)A  + (size_t)blockM * K;
    const char* Bg = (const char*)Bm + (size_t)blockN * K;
    const size_t rowB = (size_t)K;

    const int c0   = tid;
    const int row0 = c0 >> 2, col0 = (c0 & 3) << 4;
    const int row1 = (c0 + 256) >> 2, col1 = ((c0 + 256) & 3) << 4;

    auto issue_tile = [&](int kt, int buf) {
        ASYNC_COPY16(&AsB[buf][row0 * 64 + col0], Ag + (size_t)row0 * rowB + kt + col0);
        ASYNC_COPY16(&BsB[buf][row0 * 64 + col0], Bg + (size_t)row0 * rowB + kt + col0);
        ASYNC_COPY16(&AsB[buf][row1 * 64 + col1], Ag + (size_t)row1 * rowB + kt + col1);
        ASYNC_COPY16(&BsB[buf][row1 * 64 + col1], Bg + (size_t)row1 * rowB + kt + col1);
    };

    v8i acc[4][2] = {};
    const int nt = K / 64;

    issue_tile(0, 0);
    for (int it = 0; it < nt; ++it) {
        const int cur = it & 1;
        ASYNC_WAIT();
        __syncthreads();
        if (it + 1 < nt) issue_tile((it + 1) * 64, cur ^ 1);

        const char* As = AsB[cur];
        const char* Bs = BsB[cur];
        Frag8 bfr[2];
#pragma unroll
        for (int nf = 0; nf < 2; ++nf) {
            int br = wn * 32 + nf * 16 + r16;
            bfr[nf].d[0] = *(const i32x2*)(Bs + br * 64 + hi * 8 + 0);
            bfr[nf].d[1] = *(const i32x2*)(Bs + br * 64 + hi * 8 + 16);
            bfr[nf].d[2] = *(const i32x2*)(Bs + br * 64 + hi * 8 + 32);
            bfr[nf].d[3] = *(const i32x2*)(Bs + br * 64 + hi * 8 + 48);
        }
#pragma unroll
        for (int mf = 0; mf < 4; ++mf) {
            Frag8 a;
            int ar = wm * 64 + mf * 16 + r16;
            a.d[0] = *(const i32x2*)(As + ar * 64 + hi * 8 + 0);
            a.d[1] = *(const i32x2*)(As + ar * 64 + hi * 8 + 16);
            a.d[2] = *(const i32x2*)(As + ar * 64 + hi * 8 + 32);
            a.d[3] = *(const i32x2*)(As + ar * 64 + hi * 8 + 48);
#pragma unroll
            for (int nf = 0; nf < 2; ++nf) {
                acc[mf][nf] = __builtin_amdgcn_wmma_i32_16x16x64_iu8(
                    true, a.v, true, bfr[nf].v, acc[mf][nf], false, false);
            }
        }
    }

#pragma unroll
    for (int nf = 0; nf < 2; ++nf) {
        const int n = blockN + wn * 32 + nf * 16 + r16;
        const float rs = rsqrtf(var[n] + 1e-5f);
        const float s  = gam[n] * rs;
        const float t  = s * (bias[n] - mean[n]) + bet[n];
#pragma unroll
        for (int mf = 0; mf < 4; ++mf) {
            const int mbase = blockM + wm * 64 + mf * 16 + hi * 8;
#pragma unroll
            for (int r = 0; r < 8; ++r) {
                float v = s * (float)acc[mf][nf][r] + t;
                v = fminf(1.0f, fmaxf(-1.0f, v));
                if (BINOUT) {
                    out8[(size_t)(mbase + r) * N + n] =
                        (v >= 0.0f) ? (signed char)1 : (signed char)-1;
                } else {
                    outf[(size_t)(mbase + r) * N + n] = v;
                }
            }
        }
    }
}

__global__ __launch_bounds__(256) void fc4_lsm_kernel(
    const float* __restrict__ h,
    const float* __restrict__ w4,
    const float* __restrict__ b4,
    float* __restrict__ out,
    int M)
{
    __shared__ float ws[10 * 768];
    for (int i = threadIdx.x; i < 10 * 768; i += 256) ws[i] = w4[i];
    __syncthreads();

    const int row = blockIdx.x * 256 + threadIdx.x;
    if (row >= M) return;

    const float* hr = h + (size_t)row * 768;
    float acc[10];
#pragma unroll
    for (int c = 0; c < 10; ++c) acc[c] = 0.0f;

    for (int k = 0; k < 768; k += 4) {
        float x0 = hr[k + 0], x1 = hr[k + 1], x2 = hr[k + 2], x3 = hr[k + 3];
#pragma unroll
        for (int c = 0; c < 10; ++c) {
            acc[c] += x0 * ws[c * 768 + k + 0] + x1 * ws[c * 768 + k + 1]
                    + x2 * ws[c * 768 + k + 2] + x3 * ws[c * 768 + k + 3];
        }
    }

    float mx = -3.4e38f;
#pragma unroll
    for (int c = 0; c < 10; ++c) { acc[c] += b4[c]; mx = fmaxf(mx, acc[c]); }
    float sum = 0.0f;
#pragma unroll
    for (int c = 0; c < 10; ++c) sum += __expf(acc[c] - mx);
    const float lse = mx + __logf(sum);
#pragma unroll
    for (int c = 0; c < 10; ++c) out[(size_t)row * 10 + c] = acc[c] - lse;
}

extern "C" void kernel_launch(void* const* d_in, const int* in_sizes, int n_in,
                              void* d_out, int out_size, void* d_ws, size_t ws_size,
                              hipStream_t stream)
{
    (void)in_sizes; (void)n_in; (void)out_size; (void)ws_size;

    const float* x   = (const float*)d_in[0];
    const float* w1  = (const float*)d_in[1];
    const float* b1  = (const float*)d_in[2];
    const float* w2  = (const float*)d_in[3];
    const float* b2  = (const float*)d_in[4];
    const float* w3  = (const float*)d_in[5];
    const float* b3  = (const float*)d_in[6];
    const float* w4  = (const float*)d_in[7];
    const float* b4  = (const float*)d_in[8];
    const float* g1  = (const float*)d_in[9];
    const float* be1 = (const float*)d_in[10];
    const float* m1  = (const float*)d_in[11];
    const float* v1  = (const float*)d_in[12];
    const float* g2  = (const float*)d_in[13];
    const float* be2 = (const float*)d_in[14];
    const float* m2  = (const float*)d_in[15];
    const float* v2  = (const float*)d_in[16];
    const float* g3  = (const float*)d_in[17];
    const float* be3 = (const float*)d_in[18];
    const float* m3  = (const float*)d_in[19];
    const float* v3  = (const float*)d_in[20];
    float* out = (float*)d_out;

    const int B  = 16384;
    const int D  = 784;
    const int KP = 800;
    const int H1 = 3072;
    const int H2 = 1536;
    const int H3 = 768;

    char* ws = (char*)d_ws;
    size_t off = 0;
    auto take = [&](size_t bytes) -> char* {
        char* p = ws + off;
        off = (off + bytes + 255) & ~(size_t)255;
        return p;
    };
    _Float16*    xh  = (_Float16*)   take((size_t)B  * KP * 2);
    _Float16*    w1h = (_Float16*)   take((size_t)H1 * KP * 2);
    signed char* h1b = (signed char*)take((size_t)B  * H1);
    signed char* w2b = (signed char*)take((size_t)H2 * H1);
    signed char* h2b = (signed char*)take((size_t)B  * H2);
    signed char* w3b = (signed char*)take((size_t)H3 * H2);
    float*       h3  = (float*)      take((size_t)B  * H3 * 4);

    {
        int n = B * KP;
        cvt_pad_f16_kernel<<<(n + 255) / 256, 256, 0, stream>>>(x, xh, B, D, KP, 0);
    }
    {
        int n = H1 * KP;
        cvt_pad_f16_kernel<<<(n + 255) / 256, 256, 0, stream>>>(w1, w1h, H1, D, KP, 1);
    }
    {
        int n = H2 * H1;
        sign_i8_kernel<<<(n + 255) / 256, 256, 0, stream>>>(w2, w2b, n);
    }
    {
        int n = H3 * H2;
        sign_i8_kernel<<<(n + 255) / 256, 256, 0, stream>>>(w3, w3b, n);
    }

    gemm_f16_kernel<<<dim3(H1 / 128, B / 128), 256, 0, stream>>>(
        xh, w1h, b1, g1, be1, m1, v1, h1b, H1, KP);

    gemm_i8_kernel<1><<<dim3(H2 / 128, B / 128), 256, 0, stream>>>(
        h1b, w2b, b2, g2, be2, m2, v2, h2b, (float*)nullptr, H2, H1);

    gemm_i8_kernel<0><<<dim3(H3 / 128, B / 128), 256, 0, stream>>>(
        h2b, w3b, b3, g3, be3, m3, v3, (signed char*)nullptr, h3, H3, H2);

    fc4_lsm_kernel<<<B / 256, 256, 0, stream>>>(h3, w4, b4, out, B);
}
